// ReceptorEncoderGVP_33406255628289
// MI455X (gfx1250) — matrix operand order repaired and checked by reading, not compiled, not run on hardware
//
#include <hip/hip_runtime.h>

typedef __attribute__((ext_vector_type(16))) _Float16 v16h;
typedef __attribute__((ext_vector_type(8)))  float    v8f;

#define N_B    8
#define N_PER  2000
#define N_REC  (N_B*N_PER)
#define N_K    20
#define N_KP   (N_B*N_K)
#define KNN_K  8
#define SDIM   128
#define VDIM   16
#define INS    40
#define NRBF   16
#define DEG    32
#define INVZ   0.1f
#define DMAXRR 10.0f
#define DMAXRK 12.0f

// fast activations: v_rcp_f32 instead of the IEEE division ladder
__device__ __forceinline__ float dsigm(float x){ return __builtin_amdgcn_rcpf(1.f + __expf(-x)); }
__device__ __forceinline__ float dsilu(float x){ return x * dsigm(x); }

__device__ __forceinline__ v8f wmma_f16(v16h a, v16h b, v8f c){
  // v_wmma_f32_16x16x32_f16: (neg_a, A, neg_b, B, c_mod, C, reuse_a, reuse_b)
  return __builtin_amdgcn_wmma_f32_16x16x32_f16(false, a, false, b, (short)0, c, false, false);
}

// CDNA5 16-bit A-matrix layout (ISA 7.12.2): halfword e of lane (row, hi)
// holds K = (e%8) + (e>=8 ? 16 : 0) + hi*8 within a 32-K tile.
// a_fragidx inverts that: flat LDS half-index so each lane's fragment is one
// contiguous 32B v16h: index = ((kt*32 + hi*16 + row)*16 + e).
__device__ __forceinline__ int a_fragidx(int row, int k){
  int kt = k >> 5, kl = k & 31;
  int hi = (kl >> 3) & 1;
  int e  = (kl & 7) + ((kl >> 4) << 3);
  return (((kt << 5) + (hi << 4) + row) << 4) + e;
}

// ---------------------------------------------------------------------------
// Pack an f32 weight matrix W[K][N] into f16 WMMA B-fragment order:
// half index ((kt*NT+nt)*32 + lane)*16 + e holds W[kt*32 + (e%8) + 8*(lane>=16) + 16*(e>=8)][nt*16 + lane%16]
// (B layout matches A: halfword e of lane half hi holds K = (e%8) + hi*8 + (e>=8 ? 16 : 0))
// ---------------------------------------------------------------------------
__global__ void pack_b_kernel(const float* __restrict__ W, _Float16* __restrict__ out,
                              int K, int N, int KT, int NT)
{
  int t = blockIdx.x * blockDim.x + threadIdx.x;
  int total = KT * NT * 32 * 16;
  if (t >= total) return;
  int e       = t & 15;
  int lane    = (t >> 4) & 31;
  int tileIdx = t >> 9;
  int nt = tileIdx % NT, kt = tileIdx / NT;
  int n = nt * 16 + (lane & 15);
  int k = kt * 32 + (e & 7) + ((lane >> 4) << 3) + ((e >> 3) << 4);
  float v = (k < K && n < N) ? W[k * N + n] : 0.f;
  out[t] = (_Float16)v;
}

// ---------------------------------------------------------------------------
// GVP edge-message kernel. One wave per 16-edge tile.
// A row (f16, fragment-order LDS) = [s_src(128) | (s_dst(128)) | rbf(16) | vn(H) | 0..]
// Ws GEMM via WMMA f16->f32, fused bias+SiLU, scalar scatter-atomics /Z.
// Gate GEMM via WMMA, then gated vector messages scattered with atomics.
// ---------------------------------------------------------------------------
template<int SI, int VI, int H, int KPAD, bool USE_DST>
__global__ void __launch_bounds__(32) gvp_msg_kernel(
    const int* __restrict__ src, const int* __restrict__ dst, int nE,
    const float* __restrict__ sSrc, const float* __restrict__ xSrc, const float* __restrict__ vSrc,
    const float* __restrict__ sDst, const float* __restrict__ xDst, const float* __restrict__ vDst,
    const float* __restrict__ Wh, const _Float16* __restrict__ WsPk, const float* __restrict__ bs,
    const float* __restrict__ Wv, const _Float16* __restrict__ WgPk, const float* __restrict__ bg,
    float* __restrict__ agS, float* __restrict__ agV, float dmax)
{
  constexpr int KT = KPAD / 32;
  __shared__ __align__(32) _Float16 Af[KT * 512];   // A in fragment order
  __shared__ __align__(32) _Float16 SOf[4 * 512];   // silu(Ws out) in fragment order
  __shared__ float VH[16][H][3];
  __shared__ float GATE[16][16];
  __shared__ int   SIDX[16], DIDX[16], VAL[16];

  const int lane = threadIdx.x;
  const int m = lane & 15, hi = lane >> 4;
  const int tile = blockIdx.x;

  // warm L2 with the shared weight streams (global_prefetch_b8)
  __builtin_prefetch((const void*)WsPk, 0, 3);
  __builtin_prefetch((const void*)WgPk, 0, 3);

  // ---- phase 0: edge indices (lanes 0..15)
  if (hi == 0) {
    int e  = tile * 16 + m;
    int ok = (e < nE) ? 1 : 0;
    SIDX[m] = ok ? src[e] : 0;
    DIDX[m] = ok ? dst[e] : 0;
    VAL[m]  = ok;
  }
  __syncthreads();

  // ---- phase 1: geometry + Wh mix + vector norms; h-range split across lane halves
  {
    int is = SIDX[m], id = DIDX[m], ok = VAL[m];
    float dx = xSrc[is*3+0] - xDst[id*3+0];
    float dy = xSrc[is*3+1] - xDst[id*3+1];
    float dz = xSrc[is*3+2] - xDst[id*3+2];
    float dist = sqrtf(dx*dx + dy*dy + dz*dz + 1e-8f);
    float inv  = __builtin_amdgcn_rcpf(dist);

    float vin[VI][3];
    #pragma unroll
    for (int i = 0; i < VDIM; ++i) {
      vin[i][0] = vSrc[is*(VDIM*3) + i*3 + 0];
      vin[i][1] = vSrc[is*(VDIM*3) + i*3 + 1];
      vin[i][2] = vSrc[is*(VDIM*3) + i*3 + 2];
    }
    if (USE_DST) {
      #pragma unroll
      for (int i = 0; i < VDIM; ++i) {
        vin[VDIM+i][0] = vDst[id*(VDIM*3) + i*3 + 0];
        vin[VDIM+i][1] = vDst[id*(VDIM*3) + i*3 + 1];
        vin[VDIM+i][2] = vDst[id*(VDIM*3) + i*3 + 2];
      }
    }
    vin[VI-1][0] = dx*inv; vin[VI-1][1] = dy*inv; vin[VI-1][2] = dz*inv;

    constexpr int HH = (H + 1) / 2;
    const int hBeg = hi ? HH : 0;
    const int hEnd = hi ? H : HH;
    for (int h = hBeg; h < hEnd; ++h) {
      float a0 = 0.f, a1 = 0.f, a2 = 0.f;
      #pragma unroll
      for (int i = 0; i < VI; ++i) {
        float w = Wh[i*H + h];
        a0 += vin[i][0]*w; a1 += vin[i][1]*w; a2 += vin[i][2]*w;
      }
      VH[m][h][0] = a0; VH[m][h][1] = a1; VH[m][h][2] = a2;
      float vn = ok ? sqrtf(a0*a0 + a1*a1 + a2*a2 + 1e-8f) : 0.f;
      Af[a_fragidx(m, SI + h)] = (_Float16)vn;
    }
    if (hi == 0) {
      const float sig = dmax / (float)NRBF;
      #pragma unroll
      for (int j = 0; j < NRBF; ++j) {
        float mu = dmax * (float)j / (float)(NRBF - 1);
        float z  = (dist - mu) / sig;
        Af[a_fragidx(m, SI - NRBF + j)] = (_Float16)(ok ? __expf(-z*z) : 0.f);
      }
      for (int k = SI + H; k < KPAD; ++k) Af[a_fragidx(m, k)] = (_Float16)0.f;
    }
  }
  __syncthreads();

  // ---- phase 2: cooperative f16 staging of gathered scalar features (swizzled write)
  for (int t = lane; t < 16*SDIM; t += 32) {
    int r = t >> 7, c = t & 127;
    Af[a_fragidx(r, c)] = (_Float16)(VAL[r] ? sSrc[SIDX[r]*SDIM + c] : 0.f);
  }
  if (USE_DST) {
    for (int t = lane; t < 16*SDIM; t += 32) {
      int r = t >> 7, c = t & 127;
      Af[a_fragidx(r, SDIM + c)] = (_Float16)(VAL[r] ? sDst[DIDX[r]*SDIM + c] : 0.f);
    }
  }
  __syncthreads();

  // ---- phase 3: Ws GEMM (KPAD x 128) via WMMA, bias+SiLU fused, scalar scatter
  const v16h* Ap = (const v16h*)Af;
  const v16h* Bp = (const v16h*)WsPk;
  for (int nt = 0; nt < 8; ++nt) {
    v8f acc = {};
    for (int kt = 0; kt < KT; ++kt) {
      v16h a = Ap[kt*32 + lane];                 // contiguous 32B fragment
      v16h b = Bp[(kt*8 + nt)*32 + lane];
      acc = wmma_f16(a, b, acc);
    }
    #pragma unroll
    for (int r = 0; r < 8; ++r) {
      int row = r + hi*8;
      int col = nt*16 + m;
      float so = dsilu(acc[r] + bs[col]);
      SOf[a_fragidx(row, col)] = (_Float16)so;
      if (VAL[row]) atomicAdd(&agS[DIDX[row]*SDIM + col], so * INVZ);
    }
  }
  __syncthreads();

  // ---- phase 4: gate GEMM (128 -> 16) via WMMA + sigmoid
  {
    const v16h* Sp = (const v16h*)SOf;
    v8f acc = {};
    for (int kt = 0; kt < 4; ++kt) {
      v16h a = Sp[kt*32 + lane];
      v16h b = ((const v16h*)WgPk)[kt*32 + lane];
      acc = wmma_f16(a, b, acc);
    }
    #pragma unroll
    for (int r = 0; r < 8; ++r) GATE[r + hi*8][m] = dsigm(acc[r] + bg[m]);
  }
  __syncthreads();

  // ---- phase 5: gated vector messages vu = vh @ Wv, scatter
  for (int t = lane; t < 16*VDIM*3; t += 32) {
    int row = t / (VDIM*3);
    int rem = t - row*(VDIM*3);
    int o = rem / 3, c = rem - o*3;
    if (!VAL[row]) continue;
    float vu = 0.f;
    for (int h = 0; h < H; ++h) vu += VH[row][h][c] * Wv[h*VDIM + o];
    atomicAdd(&agV[DIDX[row]*(VDIM*3) + o*3 + c], vu * GATE[row][o] * INVZ);
  }
}

// ---------------------------------------------------------------------------
// GVP node-update kernel (K = 2S + 2V = 288 = 9x32). One wave per 16 nodes.
// Fuses WMMA Ws GEMM, bias+SiLU, residual, LayerNorm and gated vector update.
// ---------------------------------------------------------------------------
__global__ void __launch_bounds__(32) gvp_upd_kernel(
    int nNodes,
    const float* __restrict__ sIn, const float* __restrict__ vIn,
    const float* __restrict__ agS, const float* __restrict__ agV,
    const float* __restrict__ Wh, const _Float16* __restrict__ WsPk, const float* __restrict__ bs,
    const float* __restrict__ Wv, const _Float16* __restrict__ WgPk, const float* __restrict__ bg,
    const float* __restrict__ lnG, const float* __restrict__ lnB,
    float* __restrict__ sOut, float* __restrict__ vOut)
{
  constexpr int H = 2*VDIM, VI = 2*VDIM, KPAD = 2*SDIM + 2*VDIM, KT = KPAD/32;
  __shared__ __align__(32) _Float16 Af[KT * 512];
  __shared__ __align__(32) _Float16 SOf[4 * 512];
  __shared__ float VH[16][H][3];
  __shared__ float US[16][SDIM];
  __shared__ float GATE[16][16];
  __shared__ float MEANR[16], RSTD[16];

  const int lane = threadIdx.x, m = lane & 15, hi = lane >> 4;
  const int node0 = blockIdx.x * 16;

  __builtin_prefetch((const void*)WsPk, 0, 3);
  __builtin_prefetch((const void*)WgPk, 0, 3);

  // ---- phase 1: Wh mix + norms, h-range split across lane halves
  {
    int node = node0 + m;
    int ok = node < nNodes;
    int nd = ok ? node : 0;
    float vin[VI][3];
    #pragma unroll
    for (int i = 0; i < VDIM; ++i) {
      #pragma unroll
      for (int c = 0; c < 3; ++c) {
        vin[i][c]      = vIn[nd*(VDIM*3) + i*3 + c];
        vin[VDIM+i][c] = agV[nd*(VDIM*3) + i*3 + c];
      }
    }
    constexpr int HH = H / 2;
    const int hBeg = hi ? HH : 0;
    const int hEnd = hi ? H : HH;
    for (int h = hBeg; h < hEnd; ++h) {
      float a0 = 0.f, a1 = 0.f, a2 = 0.f;
      #pragma unroll
      for (int i = 0; i < VI; ++i) {
        float w = Wh[i*H + h];
        a0 += vin[i][0]*w; a1 += vin[i][1]*w; a2 += vin[i][2]*w;
      }
      VH[m][h][0] = a0; VH[m][h][1] = a1; VH[m][h][2] = a2;
      Af[a_fragidx(m, 2*SDIM + h)] = (_Float16)(ok ? sqrtf(a0*a0 + a1*a1 + a2*a2 + 1e-8f) : 0.f);
    }
  }
  __syncthreads();

  for (int t = lane; t < 16*SDIM; t += 32) {
    int r = t >> 7, c = t & 127;
    int node = node0 + r;
    int ok = node < nNodes; int nd = ok ? node : 0;
    Af[a_fragidx(r, c)]        = (_Float16)(ok ? sIn[nd*SDIM + c] : 0.f);
    Af[a_fragidx(r, SDIM + c)] = (_Float16)(ok ? agS[nd*SDIM + c] : 0.f);
  }
  __syncthreads();

  const v16h* Ap = (const v16h*)Af;
  const v16h* Bp = (const v16h*)WsPk;
  for (int nt = 0; nt < 8; ++nt) {
    v8f acc = {};
    for (int kt = 0; kt < KT; ++kt) {
      v16h a = Ap[kt*32 + lane];
      v16h b = Bp[(kt*8 + nt)*32 + lane];
      acc = wmma_f16(a, b, acc);
    }
    #pragma unroll
    for (int r = 0; r < 8; ++r) {
      int row = r + hi*8, col = nt*16 + m;
      int node = node0 + row;
      float us = dsilu(acc[r] + bs[col]);
      SOf[a_fragidx(row, col)] = (_Float16)us;
      US[row][col] = us + ((node < nNodes) ? sIn[node*SDIM + col] : 0.f);
    }
  }
  __syncthreads();

  {
    const v16h* Sp = (const v16h*)SOf;
    v8f acc = {};
    for (int kt = 0; kt < 4; ++kt) {
      v16h a = Sp[kt*32 + lane];
      v16h b = ((const v16h*)WgPk)[kt*32 + lane];
      acc = wmma_f16(a, b, acc);
    }
    #pragma unroll
    for (int r = 0; r < 8; ++r) GATE[r + hi*8][m] = dsigm(acc[r] + bg[m]);
  }
  __syncthreads();

  if (hi == 0) {
    float s = 0.f;
    for (int c = 0; c < SDIM; ++c) s += US[m][c];
    float mean = s / (float)SDIM;
    float v = 0.f;
    for (int c = 0; c < SDIM; ++c) { float d = US[m][c] - mean; v += d*d; }
    MEANR[m] = mean;
    RSTD[m]  = rsqrtf(v / (float)SDIM + 1e-5f);
  }
  __syncthreads();

  for (int t = lane; t < 16*SDIM; t += 32) {
    int r = t >> 7, c = t & 127;
    int node = node0 + r;
    if (node < nNodes)
      sOut[node*SDIM + c] = (US[r][c] - MEANR[r]) * RSTD[r] * lnG[c] + lnB[c];
  }
  for (int t = lane; t < 16*VDIM*3; t += 32) {
    int row = t / (VDIM*3), rem = t - row*(VDIM*3);
    int o = rem / 3, c = rem - o*3;
    int node = node0 + row;
    if (node >= nNodes) continue;
    float vu = 0.f;
    for (int h = 0; h < H; ++h) vu += VH[row][h][c] * Wv[h*VDIM + o];
    vOut[node*(VDIM*3) + o*3 + c] = vIn[node*(VDIM*3) + o*3 + c] + vu * GATE[row][o];
  }
}

// ---------------------------------------------------------------------------
// Scalar embed: silu(h0 W1 + b1) -> silu(. W2 + b2) -> LayerNorm
// ---------------------------------------------------------------------------
__global__ void __launch_bounds__(128) embed_kernel(
    const float* __restrict__ h0, const float* __restrict__ W1, const float* __restrict__ b1,
    const float* __restrict__ W2, const float* __restrict__ b2,
    const float* __restrict__ g, const float* __restrict__ bb, float* __restrict__ sOut)
{
  __shared__ float hin[INS];
  __shared__ float l1[SDIM];
  __shared__ float red[SDIM];
  int node = blockIdx.x, t = threadIdx.x;
  if (t < INS) hin[t] = h0[node*INS + t];
  __syncthreads();
  float acc = b1[t];
  for (int i = 0; i < INS; ++i) acc += hin[i] * W1[i*SDIM + t];
  l1[t] = dsilu(acc);
  __syncthreads();
  acc = b2[t];
  for (int i = 0; i < SDIM; ++i) acc += l1[i] * W2[i*SDIM + t];
  float x = dsilu(acc);
  red[t] = x; __syncthreads();
  for (int s = 64; s > 0; s >>= 1) { if (t < s) red[t] += red[t+s]; __syncthreads(); }
  float mean = red[0] / (float)SDIM; __syncthreads();
  float d = x - mean;
  red[t] = d*d; __syncthreads();
  for (int s = 64; s > 0; s >>= 1) { if (t < s) red[t] += red[t+s]; __syncthreads(); }
  float rstd = rsqrtf(red[0] / (float)SDIM + 1e-5f);
  sOut[node*SDIM + t] = d * rstd * g[t] + bb[t];
}

__global__ void __launch_bounds__(128) mean_kernel(const float* __restrict__ s,
                                                   float* __restrict__ meanF)
{
  int b = blockIdx.x, c = threadIdx.x;
  float acc = 0.f;
  const float* p = s + (size_t)b * N_PER * SDIM + c;
  for (int n = 0; n < N_PER; ++n) acc += p[(size_t)n * SDIM];
  meanF[b*SDIM + c] = acc * (1.f / (float)N_PER);
}

__global__ void __launch_bounds__(256) kpemb_kernel(
    const float* __restrict__ meanF, const float* __restrict__ W, const float* __restrict__ bias,
    const float* __restrict__ g, const float* __restrict__ bb, float* __restrict__ kpS0)
{
  constexpr int OUT = SDIM * N_K;
  __shared__ float mf[SDIM];
  __shared__ float emb[OUT];
  __shared__ float red[256];
  int b = blockIdx.x, t = threadIdx.x;
  if (t < SDIM) mf[t] = meanF[b*SDIM + t];
  __syncthreads();
  for (int o = t; o < OUT; o += 256) {
    float acc = bias[o];
    for (int i = 0; i < SDIM; ++i) acc += mf[i] * W[i*OUT + o];
    emb[o] = dsilu(acc);
  }
  __syncthreads();
  float ls = 0.f;
  for (int o = t; o < OUT; o += 256) ls += emb[o];
  red[t] = ls; __syncthreads();
  for (int s = 128; s > 0; s >>= 1) { if (t < s) red[t] += red[t+s]; __syncthreads(); }
  float mean = red[0] / (float)OUT; __syncthreads();
  float lv = 0.f;
  for (int o = t; o < OUT; o += 256) { float d = emb[o] - mean; lv += d*d; }
  red[t] = lv; __syncthreads();
  for (int s = 128; s > 0; s >>= 1) { if (t < s) red[t] += red[t+s]; __syncthreads(); }
  float rstd = rsqrtf(red[0] / (float)OUT + 1e-5f);
  for (int o = t; o < OUT; o += 256)
    kpS0[(size_t)b*OUT + o] = (emb[o] - mean) * rstd * g[o] + bb[o];
}

__global__ void __launch_bounds__(128) linear128_kernel(
    const float* __restrict__ in, const float* __restrict__ W, float* __restrict__ out)
{
  __shared__ float r[SDIM];
  int row = blockIdx.x, t = threadIdx.x;
  r[t] = in[(size_t)row*SDIM + t];
  __syncthreads();
  float acc = 0.f;
  for (int i = 0; i < SDIM; ++i) acc += r[i] * W[i*SDIM + t];
  out[(size_t)row*SDIM + t] = acc;
}

__global__ void __launch_bounds__(256) attn_kernel(
    const float* __restrict__ ftDst, const float* __restrict__ ftSrc,
    const float* __restrict__ x, float* __restrict__ kpPos)
{
  __shared__ float q[SDIM];
  __shared__ float lg[N_PER];
  __shared__ float red[256];
  int bk = blockIdx.x, b = bk / N_K, t = threadIdx.x;
  if (t < SDIM) q[t] = ftDst[bk*SDIM + t];
  __syncthreads();
  float lmax = -3.4e38f;
  for (int n = t; n < N_PER; n += 256) {
    const float* p = ftSrc + (size_t)(b*N_PER + n) * SDIM;
    float acc = 0.f;
    for (int i = 0; i < SDIM; ++i) acc += q[i] * p[i];
    acc *= 0.08838834764831845f;   // 1/sqrt(128)
    lg[n] = acc;
    lmax = fmaxf(lmax, acc);
  }
  red[t] = lmax; __syncthreads();
  for (int s = 128; s > 0; s >>= 1) { if (t < s) red[t] = fmaxf(red[t], red[t+s]); __syncthreads(); }
  float gmax = red[0]; __syncthreads();
  float ls = 0.f;
  for (int n = t; n < N_PER; n += 256) { float e = __expf(lg[n] - gmax); lg[n] = e; ls += e; }
  red[t] = ls; __syncthreads();
  for (int s = 128; s > 0; s >>= 1) { if (t < s) red[t] += red[t+s]; __syncthreads(); }
  float denom = red[0]; __syncthreads();
  float px = 0.f, py = 0.f, pz = 0.f;
  for (int n = t; n < N_PER; n += 256) {
    float w = lg[n];
    const float* xp = x + (size_t)(b*N_PER + n) * 3;
    px += w*xp[0]; py += w*xp[1]; pz += w*xp[2];
  }
  red[t] = px; __syncthreads();
  for (int s = 128; s > 0; s >>= 1) { if (t < s) red[t] += red[t+s]; __syncthreads(); }
  if (t == 0) kpPos[bk*3 + 0] = red[0] / denom;
  __syncthreads();
  red[t] = py; __syncthreads();
  for (int s = 128; s > 0; s >>= 1) { if (t < s) red[t] += red[t+s]; __syncthreads(); }
  if (t == 0) kpPos[bk*3 + 1] = red[0] / denom;
  __syncthreads();
  red[t] = pz; __syncthreads();
  for (int s = 128; s > 0; s >>= 1) { if (t < s) red[t] += red[t+s]; __syncthreads(); }
  if (t == 0) kpPos[bk*3 + 2] = red[0] / denom;
}

__global__ void __launch_bounds__(256) knn_kernel(
    const float* __restrict__ kpPos, const float* __restrict__ x, int* __restrict__ rkSrc)
{
  int bk = blockIdx.x * blockDim.x + threadIdx.x;
  if (bk >= N_KP) return;
  int b = bk / N_K;
  float kx = kpPos[bk*3+0], ky = kpPos[bk*3+1], kz = kpPos[bk*3+2];
  float bestD[KNN_K]; int bestI[KNN_K];
  #pragma unroll
  for (int j = 0; j < KNN_K; ++j) { bestD[j] = 3.4e38f; bestI[j] = 0; }
  for (int n = 0; n < N_PER; ++n) {
    const float* xp = x + (size_t)(b*N_PER + n) * 3;
    float dx = xp[0]-kx, dy = xp[1]-ky, dz = xp[2]-kz;
    float d = dx*dx + dy*dy + dz*dz;
    if (d < bestD[KNN_K-1]) {
      int j = KNN_K - 1;
      while (j > 0 && bestD[j-1] > d) { bestD[j] = bestD[j-1]; bestI[j] = bestI[j-1]; --j; }
      bestD[j] = d; bestI[j] = n;
    }
  }
  #pragma unroll
  for (int j = 0; j < KNN_K; ++j) rkSrc[bk*KNN_K + j] = b*N_PER + bestI[j];
}

__global__ void rkdst_kernel(int* __restrict__ rkDst)
{
  int e = blockIdx.x * blockDim.x + threadIdx.x;
  if (e < N_KP * KNN_K) rkDst[e] = e / KNN_K;
}

// ---------------------------------------------------------------------------
extern "C" void kernel_launch(void* const* d_in, const int* in_sizes, int n_in,
                              void* d_out, int out_size, void* d_ws, size_t ws_size,
                              hipStream_t stream)
{
  (void)in_sizes; (void)n_in; (void)out_size; (void)ws_size;
  auto F = [&](int i) -> const float* { return (const float*)d_in[i]; };
  auto I = [&](int i) -> const int*   { return (const int*)d_in[i]; };

  const float* rec_h0 = F(0);
  const float* rec_x0 = F(1);
  const int*   rr_src = I(2);
  const int*   rr_dst = I(3);
  // d_in[4]=rec_batch, d_in[5]=kp_batch: implicit in the fixed layout

  // ---- workspace bump allocator (256B aligned)
  char* wsb = (char*)d_ws;
  size_t off = 0;
  auto alloc = [&](size_t bytes) -> void* {
    void* p = wsb + off;
    off += (bytes + 255) & ~(size_t)255;
    return p;
  };

  float* sA   = (float*)alloc((size_t)N_REC * SDIM * 4);
  float* sB   = (float*)alloc((size_t)N_REC * SDIM * 4);
  float* vA   = (float*)alloc((size_t)N_REC * VDIM * 3 * 4);
  float* vB   = (float*)alloc((size_t)N_REC * VDIM * 3 * 4);
  float* agS  = (float*)alloc((size_t)N_REC * SDIM * 4);
  float* agV  = (float*)alloc((size_t)N_REC * VDIM * 3 * 4);
  float* meanF  = (float*)alloc((size_t)N_B * SDIM * 4);
  float* kpS0   = (float*)alloc((size_t)N_KP * SDIM * 4);
  float* ftSrc  = (float*)alloc((size_t)N_REC * SDIM * 4);
  float* ftDst  = (float*)alloc((size_t)N_KP * SDIM * 4);
  float* kpPos  = (float*)alloc((size_t)N_KP * 3 * 4);
  int*   rkSrc  = (int*)alloc((size_t)N_KP * KNN_K * 4);
  int*   rkDst  = (int*)alloc((size_t)N_KP * KNN_K * 4);
  float* ksA = (float*)alloc((size_t)N_KP * SDIM * 4);
  float* ksB = (float*)alloc((size_t)N_KP * SDIM * 4);
  float* kvA = (float*)alloc((size_t)N_KP * VDIM * 3 * 4);
  float* kvB = (float*)alloc((size_t)N_KP * VDIM * 3 * 4);

  // ---- param layout (recursive insertion-order flatten of params dict)
  // 6..17: embed_w1,b1,w2,b2,ln_g,ln_b, kp_emb_w,b,ln_g,ln_b, kp_src_w, kp_dst_w
  // each GVP layer = 14 arrays: msg{Wh,Ws,bs,Wv,Wg,bg} upd{Wh,Ws,bs,Wv,Wg,bg} ln_g ln_b
  const int base5[5]  = {18, 32, 46, 60, 74};        // rr0, rr1, rr2, rk0, rk1
  const int msgK5[5]  = {161, 161, 161, 161, 305};
  const int msgKT5[5] = {6, 6, 6, 6, 10};

  _Float16* msgWsPk[5]; _Float16* msgWgPk[5];
  _Float16* updWsPk[5]; _Float16* updWgPk[5];

  auto packB = [&](const float* W, _Float16* outp, int K, int N, int KT, int NT) {
    int total = KT * NT * 32 * 16;
    pack_b_kernel<<<(total + 255) / 256, 256, 0, stream>>>(W, outp, K, N, KT, NT);
  };

  for (int i = 0; i < 5; ++i) {
    msgWsPk[i] = (_Float16*)alloc((size_t)msgKT5[i] * 8 * 512 * 2);
    msgWgPk[i] = (_Float16*)alloc((size_t)4 * 1 * 512 * 2);
    updWsPk[i] = (_Float16*)alloc((size_t)9 * 8 * 512 * 2);
    updWgPk[i] = (_Float16*)alloc((size_t)4 * 1 * 512 * 2);
    packB(F(base5[i] + 1),  msgWsPk[i], msgK5[i], 128, msgKT5[i], 8);
    packB(F(base5[i] + 4),  msgWgPk[i], 128, 16, 4, 1);
    packB(F(base5[i] + 7),  updWsPk[i], 288, 128, 9, 8);
    packB(F(base5[i] + 10), updWgPk[i], 128, 16, 4, 1);
  }

  // ---- embed + zero vector features
  hipMemsetAsync(vA, 0, (size_t)N_REC * VDIM * 3 * 4, stream);
  embed_kernel<<<N_REC, 128, 0, stream>>>(rec_h0, F(6), F(7), F(8), F(9), F(10), F(11), sA);

  // ---- 3 rec-rec GVP conv layers
  float *sCur = sA, *vCur = vA, *sNxt = sB, *vNxt = vB;
  const int nEdges = N_REC * DEG;
  for (int L = 0; L < 3; ++L) {
    int base = base5[L];
    hipMemsetAsync(agS, 0, (size_t)N_REC * SDIM * 4, stream);
    hipMemsetAsync(agV, 0, (size_t)N_REC * VDIM * 3 * 4, stream);
    gvp_msg_kernel<144, 17, 17, 192, false><<<(nEdges + 15) / 16, 32, 0, stream>>>(
        rr_src, rr_dst, nEdges, sCur, rec_x0, vCur, sCur, rec_x0, vCur,
        F(base + 0), msgWsPk[L], F(base + 2), F(base + 3), msgWgPk[L], F(base + 5),
        agS, agV, DMAXRR);
    gvp_upd_kernel<<<(N_REC + 15) / 16, 32, 0, stream>>>(
        N_REC, sCur, vCur, agS, agV,
        F(base + 6), updWsPk[L], F(base + 8), F(base + 9), updWgPk[L], F(base + 11),
        F(base + 12), F(base + 13), sNxt, vNxt);
    float* ts = sCur; sCur = sNxt; sNxt = ts;
    float* tv = vCur; vCur = vNxt; vNxt = tv;
  }

  // ---- keypoint init: mean readout, embedding, attention positions, KNN
  mean_kernel<<<N_B, 128, 0, stream>>>(sCur, meanF);
  kpemb_kernel<<<N_B, 256, 0, stream>>>(meanF, F(12), F(13), F(14), F(15), kpS0);
  linear128_kernel<<<N_REC, 128, 0, stream>>>(sCur, F(16), ftSrc);
  linear128_kernel<<<N_KP, 128, 0, stream>>>(kpS0, F(17), ftDst);
  attn_kernel<<<N_KP, 256, 0, stream>>>(ftDst, ftSrc, rec_x0, kpPos);
  knn_kernel<<<1, 256, 0, stream>>>(kpPos, rec_x0, rkSrc);
  rkdst_kernel<<<(N_KP * KNN_K + 255) / 256, 256, 0, stream>>>(rkDst);

  // ---- 2 rec->kp GVP conv layers
  hipMemsetAsync(ksA, 0, (size_t)N_KP * SDIM * 4, stream);
  hipMemsetAsync(kvA, 0, (size_t)N_KP * VDIM * 3 * 4, stream);
  float *ksCur = ksA, *kvCur = kvA, *ksNxt = ksB, *kvNxt = kvB;
  const int nKpE = N_KP * KNN_K;

  {
    int base = base5[3];
    hipMemsetAsync(agS, 0, (size_t)N_KP * SDIM * 4, stream);
    hipMemsetAsync(agV, 0, (size_t)N_KP * VDIM * 3 * 4, stream);
    gvp_msg_kernel<144, 17, 17, 192, false><<<(nKpE + 15) / 16, 32, 0, stream>>>(
        rkSrc, rkDst, nKpE, sCur, rec_x0, vCur, ksCur, kpPos, kvCur,
        F(base + 0), msgWsPk[3], F(base + 2), F(base + 3), msgWgPk[3], F(base + 5),
        agS, agV, DMAXRK);
    gvp_upd_kernel<<<(N_KP + 15) / 16, 32, 0, stream>>>(
        N_KP, ksCur, kvCur, agS, agV,
        F(base + 6), updWsPk[3], F(base + 8), F(base + 9), updWgPk[3], F(base + 11),
        F(base + 12), F(base + 13), ksNxt, kvNxt);
    float* ts = ksCur; ksCur = ksNxt; ksNxt = ts;
    float* tv = kvCur; kvCur = kvNxt; kvNxt = tv;
  }
  {
    int base = base5[4];
    hipMemsetAsync(agS, 0, (size_t)N_KP * SDIM * 4, stream);
    hipMemsetAsync(agV, 0, (size_t)N_KP * VDIM * 3 * 4, stream);
    gvp_msg_kernel<272, 33, 33, 320, true><<<(nKpE + 15) / 16, 32, 0, stream>>>(
        rkSrc, rkDst, nKpE, sCur, rec_x0, vCur, ksCur, kpPos, kvCur,
        F(base + 0), msgWsPk[4], F(base + 2), F(base + 3), msgWgPk[4], F(base + 5),
        agS, agV, DMAXRK);
    gvp_upd_kernel<<<(N_KP + 15) / 16, 32, 0, stream>>>(
        N_KP, ksCur, kvCur, agS, agV,
        F(base + 6), updWsPk[4], F(base + 8), F(base + 9), updWgPk[4], F(base + 11),
        F(base + 12), F(base + 13), ksNxt, kvNxt);
  }

  // ---- outputs: kp_s [160,128], kp_v [160,16,3], kp_pos [160,3]
  float* out = (float*)d_out;
  hipMemcpyAsync(out, ksNxt, (size_t)N_KP * SDIM * 4, hipMemcpyDeviceToDevice, stream);
  hipMemcpyAsync(out + (size_t)N_KP * SDIM, kvNxt, (size_t)N_KP * VDIM * 3 * 4,
                 hipMemcpyDeviceToDevice, stream);
  hipMemcpyAsync(out + (size_t)N_KP * SDIM + (size_t)N_KP * VDIM * 3, kpPos,
                 (size_t)N_KP * 3 * 4, hipMemcpyDeviceToDevice, stream);
}
